// MambaBlock_32813550141865
// MI455X (gfx1250) — matrix operand order repaired and checked by reading, not compiled, not run on hardware
//
#include <hip/hip_runtime.h>

#define DIM   1024
#define BATCH 4
#define TLEN  4096
#define MROWS (BATCH*TLEN)   // 16384

typedef __attribute__((ext_vector_type(16))) __bf16        v16bf;
typedef __attribute__((ext_vector_type(8)))  float         v8f;
typedef __attribute__((ext_vector_type(8)))  unsigned short ushort8;

__device__ __forceinline__ unsigned short f2bf_bits(float f) {
    unsigned u = __float_as_uint(f);
    unsigned r = u + 0x7fffu + ((u >> 16) & 1u);   // round-to-nearest-even
    return (unsigned short)(r >> 16);
}
__device__ __forceinline__ float fast_rcp(float x)  { return __builtin_amdgcn_rcpf(x); }
__device__ __forceinline__ float fast_sig(float z)  { return fast_rcp(1.f + __expf(-z)); }
__device__ __forceinline__ float fast_silu(float z) { return z * fast_rcp(1.f + __expf(-z)); }
__device__ __forceinline__ float fast_tanh(float z) { return 1.f - 2.f * fast_rcp(1.f + __expf(2.f * z)); }

// ---------------------------------------------------------------------------
// 1) RMSNorm: x (M, D) fp32 -> xn (M, D) bf16.  One block per row.
// ---------------------------------------------------------------------------
__global__ __launch_bounds__(256)
void rmsnorm_kernel(const float* __restrict__ x, const float* __restrict__ w,
                    unsigned short* __restrict__ xn)
{
    __shared__ float red[256];
    const int row = blockIdx.x;
    const int tid = threadIdx.x;
    const size_t base = (size_t)row * DIM;
    float v[4], s = 0.f;
#pragma unroll
    for (int j = 0; j < 4; ++j) { v[j] = x[base + tid + j * 256]; s += v[j] * v[j]; }
    red[tid] = s; __syncthreads();
    for (int off = 128; off > 0; off >>= 1) {
        if (tid < off) red[tid] += red[tid + off];
        __syncthreads();
    }
    const float scale = rsqrtf(red[0] * (1.f / DIM) + 1e-6f);
#pragma unroll
    for (int j = 0; j < 4; ++j)
        xn[base + tid + j * 256] = f2bf_bits(v[j] * scale * w[tid + j * 256]);
}

// ---------------------------------------------------------------------------
// 2) Weight convert + transpose: w (1024, N) fp32 -> wT (N, 1024) bf16.
// ---------------------------------------------------------------------------
__global__ __launch_bounds__(256)
void wtrans_kernel(const float* __restrict__ w, unsigned short* __restrict__ wT, int N)
{
    __shared__ float tile[32][33];
    const int n0 = blockIdx.x * 32, k0 = blockIdx.y * 32;
    const int tx = threadIdx.x, ty = threadIdx.y;       // 32 x 8
#pragma unroll
    for (int j = 0; j < 4; ++j)
        tile[ty + j * 8][tx] = w[(size_t)(k0 + ty + j * 8) * N + n0 + tx];
    __syncthreads();
#pragma unroll
    for (int j = 0; j < 4; ++j)
        wT[(size_t)(n0 + ty + j * 8) * DIM + k0 + tx] = f2bf_bits(tile[tx][ty + j * 8]);
}

// ---------------------------------------------------------------------------
// 3) WMMA bf16 GEMM:  C(M,N) = A(M,1024) * BT(N,1024)^T  + epilogue.
//    Block tile 128x256, 8 waves as 2(m) x 4(n); each wave 64x64 via 4x4
//    v_wmma_f32_16x16x32_bf16 accumulators (16 ds_load_b128 : 16 wmma).
//    MODE 1: n<D -> u_pre fp32 ; n>=D -> silu(gate) fp32
//    MODE 2: sigmoid/tanh/sigmoid -> params fp16 (M, 3D)
//    MODE 3: + out_b + residual -> d_out fp32
// ---------------------------------------------------------------------------
#define LDS_STRIDE 40   // bf16 units; 80B rows keep 16B alignment, stagger banks

template <int MODE>
__global__ __launch_bounds__(256)
void gemm_kernel(const unsigned short* __restrict__ A,
                 const unsigned short* __restrict__ BT,
                 const float* __restrict__ bias, int N,
                 float* __restrict__ out0, float* __restrict__ out1,
                 _Float16* __restrict__ outh,
                 const float* __restrict__ resid, float* __restrict__ outf)
{
    __shared__ unsigned short lA[128 * LDS_STRIDE];
    __shared__ unsigned short lB[256 * LDS_STRIDE];

    const int tid  = threadIdx.x;
    const int wave = tid >> 5;
    const int lane = tid & 31;
    const int wm   = wave >> 2;          // 0..1  -> 64-row strip
    const int wn   = wave & 3;           // 0..3  -> 64-col strip
    const int l15  = lane & 15;
    const int lh   = lane >> 4;
    const int m0   = blockIdx.y * 128;
    const int n0   = blockIdx.x * 256;

    v8f acc[4][4];
#pragma unroll
    for (int i = 0; i < 4; ++i)
#pragma unroll
        for (int j = 0; j < 4; ++j) acc[i][j] = (v8f){0.f,0.f,0.f,0.f,0.f,0.f,0.f,0.f};

    for (int kt = 0; kt < 1024; kt += 32) {
        if (kt + 32 < 1024) {   // next-K-tile prefetch -> global_prefetch_b8
            __builtin_prefetch(A  + (size_t)(m0 + (tid >> 1)) * 1024 + kt + 32, 0, 3);
            __builtin_prefetch(BT + (size_t)(n0 + tid) * 1024 + kt + 32, 0, 3);
        }
        ushort8 ga[2], gb[4];
#pragma unroll
        for (int i = 0; i < 2; ++i) {   // A: 128 rows x 4 chunks = 512 / 256 thr
            const int chunk = tid * 2 + i;
            const int row = chunk >> 2, c = chunk & 3;
            ga[i] = *(const ushort8*)(A + (size_t)(m0 + row) * 1024 + kt + c * 8);
        }
#pragma unroll
        for (int i = 0; i < 4; ++i)     // B: row = tid, 4 contiguous 16B chunks
            gb[i] = *(const ushort8*)(BT + (size_t)(n0 + tid) * 1024 + kt + i * 8);
        __syncthreads();
#pragma unroll
        for (int i = 0; i < 2; ++i) {
            const int chunk = tid * 2 + i;
            const int row = chunk >> 2, c = chunk & 3;
            *(ushort8*)(lA + row * LDS_STRIDE + c * 8) = ga[i];
        }
#pragma unroll
        for (int i = 0; i < 4; ++i)
            *(ushort8*)(lB + tid * LDS_STRIDE + i * 8) = gb[i];
        __syncthreads();

        v16bf af[4], bf[4];
        // A 16x32 frag: lane l -> row (l&15); K = [kb,kb+8) and [kb+16,kb+24), kb=8*(l>>4)
#pragma unroll
        for (int i = 0; i < 4; ++i) {
            const unsigned short* p = lA + (wm * 64 + i * 16 + l15) * LDS_STRIDE + lh * 8;
            ((ushort8*)&af[i])[0] = *(const ushort8*)(p);
            ((ushort8*)&af[i])[1] = *(const ushort8*)(p + 16);
        }
        // B 32x16 frag: lane l -> col (l&15); K = [kb,kb+8) and [kb+16,kb+24), kb=8*(l>>4)
#pragma unroll
        for (int j = 0; j < 4; ++j) {
            const unsigned short* p = lB + (wn * 64 + j * 16 + l15) * LDS_STRIDE + lh * 8;
            ((ushort8*)&bf[j])[0] = *(const ushort8*)(p);
            ((ushort8*)&bf[j])[1] = *(const ushort8*)(p + 16);
        }
#pragma unroll
        for (int i = 0; i < 4; ++i)
#pragma unroll
            for (int j = 0; j < 4; ++j)
                acc[i][j] = __builtin_amdgcn_wmma_f32_16x16x32_bf16(
                    false, af[i], false, bf[j], (short)0, acc[i][j], false, false);
    }

    // Epilogue.  D frag: VGPR v, lane l -> row = v + 8*(l>>4), col = (l&15).
#pragma unroll
    for (int i = 0; i < 4; ++i) {
        const int rbase = m0 + wm * 64 + i * 16 + 8 * lh;
#pragma unroll
        for (int j = 0; j < 4; ++j) {
            const int col = n0 + wn * 64 + j * 16 + l15;
            const float bia = bias[col];
#pragma unroll
            for (int v = 0; v < 8; ++v) {
                const int m = rbase + v;
                const float z = acc[i][j][v] + bia;
                if (MODE == 1) {
                    if (col < DIM) {
                        out0[(size_t)m * DIM + col] = z;                    // u (pre-conv)
                    } else {
                        out1[(size_t)m * DIM + (col - DIM)] = fast_silu(z); // silu(gate)
                    }
                } else if (MODE == 2) {
                    float r;
                    if (col < DIM)          r = fast_sig(z);     // a
                    else if (col < 2 * DIM) r = fast_tanh(z);    // b
                    else                    r = fast_sig(z);     // c
                    outh[(size_t)m * (3 * DIM) + col] = (_Float16)r;
                } else {
                    outf[(size_t)m * DIM + col] = z + resid[(size_t)m * DIM + col];
                }
            }
        }
    }
}

// ---------------------------------------------------------------------------
// 4) Causal depthwise conv (K=4) + SiLU. Writes fp32 (scan) and bf16 (GEMM2 A).
// ---------------------------------------------------------------------------
__global__ __launch_bounds__(256)
void conv_silu_kernel(const float* __restrict__ u, const float* __restrict__ cw,
                      const float* __restrict__ cb,
                      float* __restrict__ uc, unsigned short* __restrict__ ucb)
{
    const size_t idx = (size_t)blockIdx.x * 256 + threadIdx.x;   // over M*D
    const int d = (int)(idx & (DIM - 1));
    const int t = (int)((idx >> 10) & (TLEN - 1));
    float y = cb[d];
#pragma unroll
    for (int k = 0; k < 4; ++k) {
        const int tt = t - 3 + k;
        if (tt >= 0) y += u[idx + (size_t)(k - 3) * DIM] * cw[k * DIM + d];
    }
    const float s = fast_silu(y);
    uc[idx]  = s;
    ucb[idx] = f2bf_bits(s);
}

// ---------------------------------------------------------------------------
// 5) Sequential scan over T; one thread per (b, d) channel (4096 threads).
//    state = a*state + b*u ; y = (c*state + skip*u) * silu(gate) -> bf16
// ---------------------------------------------------------------------------
__global__ __launch_bounds__(256)
void scan_kernel(const _Float16* __restrict__ P, const float* __restrict__ u,
                 const float* __restrict__ g, const float* __restrict__ skip,
                 unsigned short* __restrict__ ybf)
{
    const int idx = blockIdx.x * 256 + threadIdx.x;   // [0, B*D)
    const int b = idx >> 10, d = idx & (DIM - 1);
    const float sk = skip[d];
    float st = 0.f;
    size_t rowp = (size_t)b * TLEN * (3 * DIM) + d;
    size_t rowu = (size_t)b * TLEN * DIM + d;
    for (int t = 0; t < TLEN; ++t) {
        const float a  = (float)P[rowp];
        const float bb = (float)P[rowp + DIM];
        const float c  = (float)P[rowp + 2 * DIM];
        const float uu = u[rowu];
        st = fmaf(a, st, bb * uu);
        const float y = (c * st + sk * uu) * g[rowu];
        ybf[rowu] = f2bf_bits(y);
        rowp += 3 * DIM; rowu += DIM;
    }
}

// ---------------------------------------------------------------------------
// Host orchestration
// ---------------------------------------------------------------------------
extern "C" void kernel_launch(void* const* d_in, const int* in_sizes, int n_in,
                              void* d_out, int out_size, void* d_ws, size_t ws_size,
                              hipStream_t stream)
{
    (void)in_sizes; (void)n_in; (void)out_size; (void)ws_size;
    const float* x       = (const float*)d_in[0];
    const float* norm_w  = (const float*)d_in[1];
    const float* in_w    = (const float*)d_in[2];
    const float* in_b    = (const float*)d_in[3];
    const float* conv_w  = (const float*)d_in[4];
    const float* conv_b  = (const float*)d_in[5];
    const float* param_w = (const float*)d_in[6];
    const float* param_b = (const float*)d_in[7];
    const float* skip    = (const float*)d_in[8];
    const float* out_w   = (const float*)d_in[9];
    const float* out_b   = (const float*)d_in[10];
    float* outp = (float*)d_out;

    const size_t M = MROWS;
    size_t off = 0;
    auto alloc = [&](size_t bytes) -> void* {
        void* p = (char*)d_ws + off;
        off += (bytes + 255) & ~(size_t)255;
        return p;
    };
    unsigned short* xn      = (unsigned short*)alloc(M * DIM * 2);
    unsigned short* wInT    = (unsigned short*)alloc((size_t)2048 * 1024 * 2);
    unsigned short* wParamT = (unsigned short*)alloc((size_t)3072 * 1024 * 2);
    unsigned short* wOutT   = (unsigned short*)alloc((size_t)1024 * 1024 * 2);
    float*          u_pre   = (float*)alloc(M * DIM * 4);
    float*          gate    = (float*)alloc(M * DIM * 4);
    float*          u_conv  = (float*)alloc(M * DIM * 4);
    unsigned short* u_bf    = (unsigned short*)alloc(M * DIM * 2);
    _Float16*       params  = (_Float16*)alloc(M * 3 * DIM * 2);
    unsigned short* y_bf    = (unsigned short*)alloc(M * DIM * 2);

    // 1) RMSNorm -> bf16 A
    rmsnorm_kernel<<<(int)M, 256, 0, stream>>>(x, norm_w, xn);

    // 2) Weight convert/transpose
    wtrans_kernel<<<dim3(2048 / 32, 1024 / 32), dim3(32, 8), 0, stream>>>(in_w,    wInT,    2048);
    wtrans_kernel<<<dim3(3072 / 32, 1024 / 32), dim3(32, 8), 0, stream>>>(param_w, wParamT, 3072);
    wtrans_kernel<<<dim3(1024 / 32, 1024 / 32), dim3(32, 8), 0, stream>>>(out_w,   wOutT,   1024);

    // 3) GEMM1: proj = xn @ in_w  -> u_pre, silu(gate)
    gemm_kernel<1><<<dim3(2048 / 256, (int)(M / 128)), 256, 0, stream>>>(
        xn, wInT, in_b, 2048, u_pre, gate, nullptr, nullptr, nullptr);

    // 4) conv + silu
    conv_silu_kernel<<<(int)(M * DIM / 256), 256, 0, stream>>>(u_pre, conv_w, conv_b, u_conv, u_bf);

    // 5) GEMM2: params = u @ param_w -> activated a/b/c fp16
    gemm_kernel<2><<<dim3(3072 / 256, (int)(M / 128)), 256, 0, stream>>>(
        u_bf, wParamT, param_b, 3072, nullptr, nullptr, params, nullptr, nullptr);

    // 6) scan + gating -> y bf16
    scan_kernel<<<(BATCH * DIM) / 256, 256, 0, stream>>>(params, u_conv, gate, skip, y_bf);

    // 7) GEMM3: out = y @ out_w + out_b + residual
    gemm_kernel<3><<<dim3(1024 / 256, (int)(M / 128)), 256, 0, stream>>>(
        y_bf, wOutT, out_b, 1024, nullptr, nullptr, nullptr, x, outp);
}
